// GridTaggingMatcher_46935402610848
// MI455X (gfx1250) — matrix operand order repaired and checked by reading, not compiled, not run on hardware
//
#include <hip/hip_runtime.h>
#include <math.h>

#define Bsz 2
#define Ssz 128
#define Hsz 768
#define Rsz 8
#define KS  24            // Hsz / 32 (K steps per WMMA chain)
#define NT_H 48           // Hsz / 16 (N tiles)
#define TPB 256
#define UST 388           // padded LDS row stride in uints (384+4, bank-conflict free)

typedef __attribute__((ext_vector_type(16))) __bf16          v16bf;
typedef __attribute__((ext_vector_type(16))) unsigned short  v16us;
typedef __attribute__((ext_vector_type(8)))  unsigned int    v8u;
typedef __attribute__((ext_vector_type(8)))  float           v8f;

__device__ __forceinline__ unsigned short f2bf(float x) {
  union { float f; unsigned int u; } v; v.f = x;
  unsigned int u = v.u;
  return (unsigned short)((u + 0x7fffu + ((u >> 16) & 1u)) >> 16);   // RNE
}

__device__ __forceinline__ unsigned int pk2bf(float a, float b) {
  return (unsigned int)f2bf(a) | ((unsigned int)f2bf(b) << 16);
}

// packed bf16 multiply: one V_PK_MUL_BF16 (CDNA5 VOP3P op 42)
__device__ __forceinline__ unsigned int pkmul_bf16(unsigned int a, unsigned int b) {
  unsigned int d;
  asm("v_pk_mul_bf16 %0, %1, %2" : "=v"(d) : "v"(a), "v"(b));
  return d;
}

__device__ __forceinline__ v8f wmma_bf16_u(const v8u& a, const v8u& b, v8f c) {
  return __builtin_amdgcn_wmma_f32_16x16x32_bf16(
      false, __builtin_bit_cast(v16bf, a),
      false, __builtin_bit_cast(v16bf, b),
      (short)0, c, false, false);
}

__device__ __forceinline__ v8f zero8() {
  v8f z;
#pragma unroll
  for (int i = 0; i < 8; ++i) z[i] = 0.0f;
  return z;
}

__device__ __forceinline__ float sigm(float x) { return 1.0f / (1.0f + expf(-x)); }

// ---------------- K0: mask ----------------
__global__ void k_mask(const float* __restrict__ hs, const float* __restrict__ mask,
                       float* __restrict__ hsm, int n) {
  int t = blockIdx.x * blockDim.x + threadIdx.x;
  if (t < n) hsm[t] = hs[t] * mask[t / Hsz];
}

// ---------------- K1: pack f32 weight [768 x ncols] into bf16 fragment-major ----------------
// Pack[((nt*KS+ks)*32 + lane)*16 + e] = bf16( W[(ks*32 + (lane>=16?8:0) + e + (e>=8?8:0)) * ncols + nt*16 + lane%16] )
__global__ void k_pack(const float* __restrict__ src, int ncols,
                       unsigned short* __restrict__ dst, int total) {
  int t = blockIdx.x * blockDim.x + threadIdx.x;
  if (t >= total) return;
  int lane = t & 31;
  int ks   = (t >> 5) % KS;
  int nt   = t / (KS * 32);
  int kbase = ks * 32 + ((lane >= 16) ? 8 : 0);
  int col   = nt * 16 + (lane & 15);
  unsigned int* d = (unsigned int*)(dst + (size_t)t * 16);
#pragma unroll
  for (int e = 0; e < 8; ++e) {
    float lo = src[(size_t)(kbase + 2 * e + (e >= 4 ? 8 : 0)) * ncols + col];
    float hi = src[(size_t)(kbase + 2 * e + (e >= 4 ? 8 : 0) + 1) * ncols + col];
    d[e] = pk2bf(lo, hi);
  }
}

// ---------------- K2: generic WMMA GEMM: out[M,N] (=/+=) A[M,768] x Wpack, +bias, relu ----------------
// A staged in LDS pre-converted to packed bf16: A-fragment build = 2 LDS b128 loads, no VALU.
template <int NT_TOTAL>
__global__ __launch_bounds__(TPB) void k_gemm(
    const float* __restrict__ A, const unsigned short* __restrict__ pack,
    float* __restrict__ out, const float* __restrict__ bias, int relu, int accum) {
  __shared__ unsigned int sAU[16 * UST];
  const int tid = threadIdx.x;
  const int m0  = blockIdx.x * 16;
  for (int idx = tid; idx < 16 * (Hsz / 2); idx += TPB) {
    int r = idx / (Hsz / 2), c = idx % (Hsz / 2);
    float2 f = *(const float2*)&A[(size_t)(m0 + r) * Hsz + 2 * c];
    sAU[r * UST + c] = pk2bf(f.x, f.y);
  }
  __syncthreads();

  const int wave = tid >> 5, lane = tid & 31;
  const int lm = lane & 15, hi = lane >> 4;
  constexpr int NTL = (NT_TOTAL + 7) / 8;
  v8f C[NTL];
#pragma unroll
  for (int q = 0; q < NTL; ++q) C[q] = zero8();

  const v8u* pv = (const v8u*)pack;
  for (int ks = 0; ks < KS; ++ks) {
    const int u0 = ks * 16 + hi * 4;           // uint offset: k-run start / 2
    const unsigned int* ar = &sAU[lm * UST];
    uint4 q0 = *(const uint4*)&ar[u0];
    uint4 q1 = *(const uint4*)&ar[u0 + 8];
    v8u af;
    af[0] = q0.x; af[1] = q0.y; af[2] = q0.z; af[3] = q0.w;
    af[4] = q1.x; af[5] = q1.y; af[6] = q1.z; af[7] = q1.w;
#pragma unroll
    for (int q = 0; q < NTL; ++q) {
      int nt = wave + q * 8;
      if (nt < NT_TOTAL) {
        v8u bf_ = pv[(size_t)(nt * KS + ks) * 32 + lane];
        C[q] = wmma_bf16_u(af, bf_, C[q]);
      }
    }
  }

  constexpr int ldO = NT_TOTAL * 16;
#pragma unroll
  for (int q = 0; q < NTL; ++q) {
    int nt = wave + q * 8;
    if (nt < NT_TOTAL) {
      int n = nt * 16 + lm;
      float bv = bias ? bias[n] : 0.0f;
#pragma unroll
      for (int v = 0; v < 8; ++v) {
        int m = v + 8 * hi;
        size_t idx = (size_t)(m0 + m) * ldO + n;
        float val = C[q][v];
        if (accum) val += out[idx];
        val += bv;
        if (relu) val = fmaxf(val, 0.0f);
        out[idx] = val;
      }
    }
  }
}

// ---------------- K3: fused pair-grid kernel ----------------
// block = (b, i, jt). h1[m,n] = relu(ti[i,n] + tj[j0+m,n] + sum_k hs_i[k]*hs_j[k]*W3[k,n] + b1[n])
// logits[m,r] = sum_n h1[m,n]*w2[n,r]; rel = sigmoid(logits + b2 + grid)
// hs rows staged in LDS as packed bf16; A-fragment = 4 LDS b128 loads + 8 v_pk_mul_bf16.
__global__ __launch_bounds__(TPB) void k_pair(
    const float* __restrict__ hsm, const float* __restrict__ ti, const float* __restrict__ tj,
    const unsigned short* __restrict__ pack, const float* __restrict__ b1,
    const float* __restrict__ w2, const float* __restrict__ b2,
    const float* __restrict__ grid1, const float* __restrict__ grid2,
    float* __restrict__ Spart, int* __restrict__ anyPart, int* __restrict__ priorPart) {
  __shared__ unsigned int sHsiU[Hsz / 2];
  __shared__ unsigned int sHsjU[16 * UST];
  __shared__ float sPartL[8][16][8];
  __shared__ float sVals[16][8];
  __shared__ int sAny, sPrior;

  const int tid = threadIdx.x;
  const int blk = blockIdx.x;
  const int b   = blk / (Ssz * 8);
  const int rem = blk % (Ssz * 8);
  const int i   = rem >> 3;
  const int jt  = rem & 7;
  const int j0  = jt * 16;
  const int bi  = b * Ssz + i;

  if (tid == 0) { sAny = 0; sPrior = 0; }
  for (int idx = tid; idx < Hsz / 2; idx += TPB) {
    float2 f = *(const float2*)&hsm[(size_t)bi * Hsz + 2 * idx];
    sHsiU[idx] = pk2bf(f.x, f.y);
  }
  for (int idx = tid; idx < 16 * (Hsz / 2); idx += TPB) {
    int r = idx / (Hsz / 2), c = idx % (Hsz / 2);
    float2 f = *(const float2*)&hsm[(size_t)(b * Ssz + j0 + r) * Hsz + 2 * c];
    sHsjU[r * UST + c] = pk2bf(f.x, f.y);
  }
  __syncthreads();

  const int wave = tid >> 5, lane = tid & 31;
  const int lm = lane & 15, hi = lane >> 4;
  const v8u* pv = (const v8u*)pack;

  v8f C[6];
#pragma unroll
  for (int q = 0; q < 6; ++q) C[q] = zero8();

  for (int ks = 0; ks < KS; ++ks) {
    const int u0 = ks * 16 + hi * 4;
    uint4 xi0 = *(const uint4*)&sHsiU[u0];
    uint4 xi1 = *(const uint4*)&sHsiU[u0 + 8];
    const unsigned int* jr = &sHsjU[lm * UST];
    uint4 xj0 = *(const uint4*)&jr[u0];
    uint4 xj1 = *(const uint4*)&jr[u0 + 8];
    v8u af;
    af[0] = pkmul_bf16(xi0.x, xj0.x);
    af[1] = pkmul_bf16(xi0.y, xj0.y);
    af[2] = pkmul_bf16(xi0.z, xj0.z);
    af[3] = pkmul_bf16(xi0.w, xj0.w);
    af[4] = pkmul_bf16(xi1.x, xj1.x);
    af[5] = pkmul_bf16(xi1.y, xj1.y);
    af[6] = pkmul_bf16(xi1.z, xj1.z);
    af[7] = pkmul_bf16(xi1.w, xj1.w);
#pragma unroll
    for (int q = 0; q < 6; ++q) {
      int nt = wave + q * 8;
      v8u bf_ = pv[(size_t)(nt * KS + ks) * 32 + lane];
      C[q] = wmma_bf16_u(af, bf_, C[q]);
    }
  }

  // epilogue: relu + H->8 projection, accumulated per-lane over this wave's n columns
  float acc2[8][8];
#pragma unroll
  for (int v = 0; v < 8; ++v)
#pragma unroll
    for (int r = 0; r < 8; ++r) acc2[v][r] = 0.0f;

#pragma unroll
  for (int q = 0; q < 6; ++q) {
    int n = (wave + q * 8) * 16 + lm;
    float tiv = ti[(size_t)bi * Hsz + n];
    float b1v = b1[n];
    float4 wa = *(const float4*)&w2[(size_t)n * 8];
    float4 wb = *(const float4*)&w2[(size_t)n * 8 + 4];
    float wr[8] = {wa.x, wa.y, wa.z, wa.w, wb.x, wb.y, wb.z, wb.w};
#pragma unroll
    for (int v = 0; v < 8; ++v) {
      int m = v + 8 * hi;
      float tjv = tj[(size_t)(b * Ssz + j0 + m) * Hsz + n];
      float h1 = C[q][v] + tiv + tjv + b1v;
      h1 = fmaxf(h1, 0.0f);
#pragma unroll
      for (int r = 0; r < 8; ++r) acc2[v][r] = fmaf(h1, wr[r], acc2[v][r]);
    }
  }

  // deterministic reduction over the 16 n-lanes of each half
#pragma unroll
  for (int off = 1; off <= 8; off <<= 1) {
#pragma unroll
    for (int v = 0; v < 8; ++v)
#pragma unroll
      for (int r = 0; r < 8; ++r)
        acc2[v][r] += __shfl_xor(acc2[v][r], off, 32);
  }
  if (lm == 0) {
#pragma unroll
    for (int v = 0; v < 8; ++v)
#pragma unroll
      for (int r = 0; r < 8; ++r)
        sPartL[wave][v + 8 * hi][r] = acc2[v][r];
  }
  __syncthreads();

  if (tid < 128) {
    int m = tid >> 3, r = tid & 7;
    float ls = 0.0f;
#pragma unroll
    for (int w = 0; w < 8; ++w) ls += sPartL[w][m][r];
    int j = j0 + m;
    float base = ls + b2[r];
    float v1 = sigm(base + grid1[((size_t)i * Ssz + j) * Rsz + r]);
    sVals[m][r] = v1;
    if (grid2) {
      float v2 = sigm(base + grid2[((size_t)i * Ssz + j) * Rsz + r]);
      if (v2 > 0.5f) {
        atomicOr(&sAny, 1);
        if (j < i) atomicOr(&sPrior, 1);
      }
    }
  }
  __syncthreads();
  if (tid < 8) {
    float s = 0.0f;
#pragma unroll
    for (int m = 0; m < 16; ++m) s += sVals[m][tid];
    Spart[((size_t)bi * 8 + jt) * 8 + tid] = s;
  }
  if (tid == 8 && anyPart) {
    anyPart[bi * 8 + jt]   = sAny;
    priorPart[bi * 8 + jt] = sPrior;
  }
}

// ---------------- K4a: reduce partials ----------------
__global__ void k_reduce(const float* __restrict__ SaaP, const float* __restrict__ SooP,
                         const int* __restrict__ anyP, const int* __restrict__ priorP,
                         float* __restrict__ Saa, float* __restrict__ Soo,
                         int* __restrict__ anyF, int* __restrict__ priorF) {
  int t = blockIdx.x * blockDim.x + threadIdx.x;
  if (t < 2048) {
    int bi = t >> 3, r = t & 7;
    float s = 0.0f;
    for (int jt = 0; jt < 8; ++jt) s += SaaP[bi * 64 + jt * 8 + r];
    Saa[t] = s;
  } else if (t < 4096) {
    int tt = t - 2048, bi = tt >> 3, r = tt & 7;
    float s = 0.0f;
    for (int jt = 0; jt < 8; ++jt) s += SooP[bi * 64 + jt * 8 + r];
    Soo[tt] = s;
  } else if (t < 4096 + 256) {
    int bi = t - 4096;
    int a = 0, p = 0;
    for (int jt = 0; jt < 8; ++jt) { a |= anyP[bi * 8 + jt]; p |= priorP[bi * 8 + jt]; }
    anyF[bi] = a; priorF[bi] = p;
  }
}

// ---------------- K4b: ctx = (Saa @ rel_emb)/S ----------------
__global__ void k_ctx(const float* __restrict__ Saa, const float* __restrict__ Soo,
                      const float* __restrict__ rel_emb,
                      float* __restrict__ ctx_a, float* __restrict__ ctx_o, int n) {
  int t = blockIdx.x * blockDim.x + threadIdx.x;
  if (t >= n) return;
  int bi = t / Hsz, h = t % Hsz;
  float a = 0.0f, o = 0.0f;
#pragma unroll
  for (int r = 0; r < 8; ++r) {
    float e = rel_emb[r * Hsz + h];
    a += Saa[bi * 8 + r] * e;
    o += Soo[bi * 8 + r] * e;
  }
  ctx_a[t] = a * (1.0f / Ssz);
  ctx_o[t] = o * (1.0f / Ssz);
}

// ---------------- K6: layer3 + consistency MLP + sequential causality ----------------
__global__ __launch_bounds__(TPB) void k_final(
    const float* __restrict__ h2a, const float* __restrict__ h2o,
    const float* __restrict__ adw3, const float* __restrict__ adb3,
    const float* __restrict__ odw3, const float* __restrict__ odb3,
    const float* __restrict__ ccw1, const float* __restrict__ ccb1,
    const float* __restrict__ ccw2, const float* __restrict__ ccb2,
    const float* __restrict__ ccw3, const float* __restrict__ ccb3,
    const int* __restrict__ expA, const int* __restrict__ expO,
    const int* __restrict__ anyF, const int* __restrict__ priorF,
    float* __restrict__ outp) {
  __shared__ float A[Bsz * Ssz][4];
  __shared__ float O[Bsz * Ssz][4];
  __shared__ float sc1[Bsz * Ssz], s2a[Bsz * Ssz], s2o[Bsz * Ssz], s4v[Bsz * Ssz];
  const int t = threadIdx.x;
  const int bi = t;
  const int b = bi / Ssz, ip = bi % Ssz;

  float asp[4], opi[4];
#pragma unroll
  for (int c = 0; c < 4; ++c) {
    float sa = adb3[c], so = odb3[c];
    for (int k = 0; k < 64; ++k) {
      sa += h2a[bi * 64 + k] * adw3[k * 4 + c];
      so += h2o[bi * 64 + k] * odw3[k * 4 + c];
    }
    asp[c] = sa; opi[c] = so;
  }
  // consistency MLP
  float comb[8] = {asp[0], asp[1], asp[2], asp[3], opi[0], opi[1], opi[2], opi[3]};
  float hh[32];
  for (int o2 = 0; o2 < 32; ++o2) {
    float s = ccb1[o2];
#pragma unroll
    for (int k = 0; k < 8; ++k) s += comb[k] * ccw1[k * 32 + o2];
    hh[o2] = fmaxf(s, 0.0f);
  }
  float h2c[16];
  for (int o2 = 0; o2 < 16; ++o2) {
    float s = ccb2[o2];
    for (int k = 0; k < 32; ++k) s += hh[k] * ccw2[k * 16 + o2];
    h2c[o2] = fmaxf(s, 0.0f);
  }
  float sc = ccb3[0];
  for (int k = 0; k < 16; ++k) sc += h2c[k] * ccw3[k];
  float score = sigm(sc);
  float scale1 = (score < 0.5f) ? 2.0f * score : 1.0f;
  // softmax probs (first two classes)
  float mA = fmaxf(fmaxf(asp[0], asp[1]), fmaxf(asp[2], asp[3]));
  float ea0 = expf(asp[0] - mA), ea1 = expf(asp[1] - mA), ea2 = expf(asp[2] - mA), ea3 = expf(asp[3] - mA);
  float pA = (ea0 + ea1) / (ea0 + ea1 + ea2 + ea3);
  float mO = fmaxf(fmaxf(opi[0], opi[1]), fmaxf(opi[2], opi[3]));
  float eo0 = expf(opi[0] - mO), eo1 = expf(opi[1] - mO), eo2 = expf(opi[2] - mO), eo3 = expf(opi[3] - mO);
  float pO = (eo0 + eo1) / (eo0 + eo1 + eo2 + eo3);
  // window flags
  bool nearO = false, nearA = false;
  for (int d = -3; d <= 3; ++d) {
    int jj = ip + d;
    if (jj >= 0 && jj < Ssz) {
      if (expO[b * Ssz + jj] > 0) nearO = true;
      if (expA[b * Ssz + jj] > 0) nearA = true;
    }
  }
  float v2a = (pA > 0.5f && !nearO) ? 0.3f : 1.0f;
  float v2o = (pO > 0.5f && !nearA) ? 0.3f : 1.0f;
  float v4  = (anyF[bi] && priorF[bi]) ? 0.7f : 1.0f;
#pragma unroll
  for (int c = 0; c < 4; ++c) { A[bi][c] = asp[c]; O[bi][c] = opi[c]; }
  sc1[bi] = scale1; s2a[bi] = v2a; s2o[bi] = v2o; s4v[bi] = v4;
  __syncthreads();

  if (t < Bsz) {
    const int bb = t;
    for (int i2 = 0; i2 < Ssz; ++i2) {
      int idx = bb * Ssz + i2;
#pragma unroll
      for (int c = 0; c < 4; ++c) {
        float av = A[idx][c] * sc1[idx];
        float ov = O[idx][c] * sc1[idx];
        if (c < 2) { av *= s2a[idx]; ov *= s2o[idx]; }
        A[idx][c] = av; O[idx][c] = ov;
      }
      bool act = false;
      const int offs[4] = {-2, -1, 1, 2};
#pragma unroll
      for (int dd = 0; dd < 4; ++dd) {
        int jj = i2 + offs[dd];
        if (jj >= 0 && jj < Ssz) {
          int ix = bb * Ssz + jj;
          float mxA = fmaxf(fmaxf(A[ix][0], A[ix][1]), fmaxf(A[ix][2], A[ix][3]));
          float mxO = fmaxf(fmaxf(O[ix][0], O[ix][1]), fmaxf(O[ix][2], O[ix][3]));
          if (mxA > 0.5f || mxO > 0.5f) act = true;
        }
      }
      float iso = act ? 1.0f : 0.1f;
      A[idx][0] *= iso; A[idx][1] *= iso;
      O[idx][0] *= iso; O[idx][1] *= iso;
      A[idx][0] *= s4v[idx]; A[idx][1] *= s4v[idx];
    }
  }
  __syncthreads();
#pragma unroll
  for (int c = 0; c < 4; ++c) {
    outp[bi * 8 + c]     = A[bi][c];
    outp[bi * 8 + 4 + c] = O[bi][c];
  }
}

// ---------------- host side ----------------
extern "C" void kernel_launch(void* const* d_in, const int* in_sizes, int n_in,
                              void* d_out, int out_size, void* d_ws, size_t ws_size,
                              hipStream_t stream) {
  (void)in_sizes; (void)n_in; (void)out_size; (void)ws_size;
  const float* hidden  = (const float*)d_in[0];
  const float* mask    = (const float*)d_in[1];
  const int*   expA    = (const int*)d_in[2];
  const int*   expO    = (const int*)d_in[3];
  const float* aa_grid = (const float*)d_in[4];
  const float* ao_grid = (const float*)d_in[5];
  const float* oo_grid = (const float*)d_in[6];
  const float* rel_emb = (const float*)d_in[7];
  const float* ag_w1 = (const float*)d_in[8];
  const float* ag_b1 = (const float*)d_in[9];
  const float* ag_w2 = (const float*)d_in[10];
  const float* ag_b2 = (const float*)d_in[11];
  const float* og_w1 = (const float*)d_in[12];
  const float* og_b1 = (const float*)d_in[13];
  const float* og_w2 = (const float*)d_in[14];
  const float* og_b2 = (const float*)d_in[15];
  const float* ad_w1 = (const float*)d_in[16];
  const float* ad_b1 = (const float*)d_in[17];
  const float* ad_w2 = (const float*)d_in[18];
  const float* ad_b2 = (const float*)d_in[19];
  const float* ad_w3 = (const float*)d_in[20];
  const float* ad_b3 = (const float*)d_in[21];
  const float* od_w1 = (const float*)d_in[22];
  const float* od_b1 = (const float*)d_in[23];
  const float* od_w2 = (const float*)d_in[24];
  const float* od_b2 = (const float*)d_in[25];
  const float* od_w3 = (const float*)d_in[26];
  const float* od_b3 = (const float*)d_in[27];
  const float* cc_w1 = (const float*)d_in[28];
  const float* cc_b1 = (const float*)d_in[29];
  const float* cc_w2 = (const float*)d_in[30];
  const float* cc_b2 = (const float*)d_in[31];
  const float* cc_w3 = (const float*)d_in[32];
  const float* cc_b3 = (const float*)d_in[33];

  char* base = (char*)d_ws;
  size_t off = 0;
  auto carve = [&](size_t bytes) -> char* {
    off = (off + 255) & ~(size_t)255;
    char* p = base + off;
    off += bytes;
    return p;
  };
  const size_t BSH = (size_t)Bsz * Ssz * Hsz * 4;       // 786432
  const size_t PK768 = (size_t)768 * 768 * 2;           // 1179648
  const size_t PK64  = (size_t)768 * 64 * 2;            // 98304

  float* hsm  = (float*)carve(BSH);
  float* tiag = (float*)carve(BSH);
  float* tjag = (float*)carve(BSH);
  float* tiog = (float*)carve(BSH);
  float* tjog = (float*)carve(BSH);
  float* ctxa = (float*)carve(BSH);
  float* ctxo = (float*)carve(BSH);
  float* h1a  = (float*)carve(BSH);
  float* h1o  = (float*)carve(BSH);
  float* h2a  = (float*)carve((size_t)Bsz * Ssz * 64 * 4);
  float* h2o  = (float*)carve((size_t)Bsz * Ssz * 64 * 4);
  unsigned short* Pagti = (unsigned short*)carve(PK768);
  unsigned short* Pagtj = (unsigned short*)carve(PK768);
  unsigned short* Pagw3 = (unsigned short*)carve(PK768);
  unsigned short* Pogti = (unsigned short*)carve(PK768);
  unsigned short* Pogtj = (unsigned short*)carve(PK768);
  unsigned short* Pogw3 = (unsigned short*)carve(PK768);
  unsigned short* Padh  = (unsigned short*)carve(PK768);
  unsigned short* Padc  = (unsigned short*)carve(PK768);
  unsigned short* Podh  = (unsigned short*)carve(PK768);
  unsigned short* Podc  = (unsigned short*)carve(PK768);
  unsigned short* Pad2  = (unsigned short*)carve(PK64);
  unsigned short* Pod2  = (unsigned short*)carve(PK64);
  float* SaaP = (float*)carve((size_t)Bsz * Ssz * 64 * 4);
  float* SooP = (float*)carve((size_t)Bsz * Ssz * 64 * 4);
  float* Saa  = (float*)carve((size_t)Bsz * Ssz * 8 * 4);
  float* Soo  = (float*)carve((size_t)Bsz * Ssz * 8 * 4);
  int* anyP   = (int*)carve((size_t)Bsz * Ssz * 8 * 4);
  int* priorP = (int*)carve((size_t)Bsz * Ssz * 8 * 4);
  int* anyF   = (int*)carve((size_t)Bsz * Ssz * 4);
  int* priorF = (int*)carve((size_t)Bsz * Ssz * 4);

  // 0) masked hidden states
  k_mask<<<768, TPB, 0, stream>>>(hidden, mask, hsm, Bsz * Ssz * Hsz);

  // 1) weight packing
  const int tot768 = NT_H * KS * 32;           // 36864
  const int bl768  = (tot768 + TPB - 1) / TPB; // 144
  const int tot64  = 4 * KS * 32;              // 3072
  const int bl64   = (tot64 + TPB - 1) / TPB;  // 12
  k_pack<<<bl768, TPB, 0, stream>>>(ag_w1,               Hsz, Pagti, tot768);
  k_pack<<<bl768, TPB, 0, stream>>>(ag_w1 + 768 * 768,   Hsz, Pagtj, tot768);
  k_pack<<<bl768, TPB, 0, stream>>>(ag_w1 + 1536 * 768,  Hsz, Pagw3, tot768);
  k_pack<<<bl768, TPB, 0, stream>>>(og_w1,               Hsz, Pogti, tot768);
  k_pack<<<bl768, TPB, 0, stream>>>(og_w1 + 768 * 768,   Hsz, Pogtj, tot768);
  k_pack<<<bl768, TPB, 0, stream>>>(og_w1 + 1536 * 768,  Hsz, Pogw3, tot768);
  k_pack<<<bl768, TPB, 0, stream>>>(ad_w1,               Hsz, Padh,  tot768);
  k_pack<<<bl768, TPB, 0, stream>>>(ad_w1 + 768 * 768,   Hsz, Padc,  tot768);
  k_pack<<<bl768, TPB, 0, stream>>>(od_w1,               Hsz, Podh,  tot768);
  k_pack<<<bl768, TPB, 0, stream>>>(od_w1 + 768 * 768,   Hsz, Podc,  tot768);
  k_pack<<<bl64,  TPB, 0, stream>>>(ad_w2, 64, Pad2, tot64);
  k_pack<<<bl64,  TPB, 0, stream>>>(od_w2, 64, Pod2, tot64);

  // 2) ti/tj projections (M = B*S = 256 rows)
  k_gemm<48><<<16, TPB, 0, stream>>>(hsm, Pagti, tiag, nullptr, 0, 0);
  k_gemm<48><<<16, TPB, 0, stream>>>(hsm, Pagtj, tjag, nullptr, 0, 0);
  k_gemm<48><<<16, TPB, 0, stream>>>(hsm, Pogti, tiog, nullptr, 0, 0);
  k_gemm<48><<<16, TPB, 0, stream>>>(hsm, Pogtj, tjog, nullptr, 0, 0);

  // 3) fused pair-grid kernels (the big WMMA work)
  k_pair<<<Bsz * Ssz * 8, TPB, 0, stream>>>(hsm, tiag, tjag, Pagw3, ag_b1, ag_w2, ag_b2,
                                            aa_grid, ao_grid, SaaP, anyP, priorP);
  k_pair<<<Bsz * Ssz * 8, TPB, 0, stream>>>(hsm, tiog, tjog, Pogw3, og_b1, og_w2, og_b2,
                                            oo_grid, nullptr, SooP, nullptr, nullptr);

  // 4) reductions + context
  k_reduce<<<17, TPB, 0, stream>>>(SaaP, SooP, anyP, priorP, Saa, Soo, anyF, priorF);
  k_ctx<<<768, TPB, 0, stream>>>(Saa, Soo, rel_emb, ctxa, ctxo, Bsz * Ssz * Hsz);

  // 5) detect MLPs
  k_gemm<48><<<16, TPB, 0, stream>>>(hsm,  Padh, h1a, nullptr, 0, 0);
  k_gemm<48><<<16, TPB, 0, stream>>>(ctxa, Padc, h1a, ad_b1,   1, 1);
  k_gemm<4><<<16, TPB, 0, stream>>>(h1a, Pad2, h2a, ad_b2, 1, 0);
  k_gemm<48><<<16, TPB, 0, stream>>>(hsm,  Podh, h1o, nullptr, 0, 0);
  k_gemm<48><<<16, TPB, 0, stream>>>(ctxo, Podc, h1o, od_b1,   1, 1);
  k_gemm<4><<<16, TPB, 0, stream>>>(h1o, Pod2, h2o, od_b2, 1, 0);

  // 6) final: layer3 + consistency + sequential causality + output
  k_final<<<1, TPB, 0, stream>>>(h2a, h2o, ad_w3, ad_b3, od_w3, od_b3,
                                 cc_w1, cc_b1, cc_w2, cc_b2, cc_w3, cc_b3,
                                 expA, expO, anyF, priorF, (float*)d_out);
}
